// GATv2_60722247631719
// MI455X (gfx1250) — matrix operand order repaired and checked by reading, not compiled, not run on hardware
//
#include <hip/hip_runtime.h>
#include <hip/hip_bf16.h>

// GAT fused attention for MI455X (gfx1250, wave32, WMMA bf16 + TDM staging).
// B=8, N=2048, Fin=Fout=256 (matches reference).

typedef __bf16        v16bf __attribute__((ext_vector_type(16)));
typedef float         v8f   __attribute__((ext_vector_type(8)));
typedef unsigned int  v8u   __attribute__((ext_vector_type(8)));

#define NEG_SLOPE 0.2f

static constexpr int BATCH = 8;
static constexpr int NN    = 2048;        // nodes
static constexpr int FF    = 256;         // features
static constexpr int BN    = BATCH * NN;  // 16384 rows
static constexpr int HSTR  = 34;          // P-buffer stride (17-bank stride)
static constexpr int TSTR  = 40;          // transposed h slab stride in bf16 (80B rows: conflict-free b128)
static constexpr int NT    = NN / 32;     // j tiles

#if __has_builtin(__builtin_amdgcn_tensor_load_to_lds)
#define GAT_TDM 1
#else
#define GAT_TDM 0
#endif

// ---------- bf16 helpers (pure bit ops) ----------
static __device__ __forceinline__ unsigned short f2bfu(float f) {
  unsigned u = __builtin_bit_cast(unsigned, f);
  u += 0x7fffu + ((u >> 16) & 1u);      // RNE
  return (unsigned short)(u >> 16);
}
static __device__ __forceinline__ float bfu2f(unsigned short s) {
  unsigned u = ((unsigned)s) << 16;
  return __builtin_bit_cast(float, u);
}
static __device__ __forceinline__ unsigned packbf(float lo, float hi) {
  return (unsigned)f2bfu(lo) | ((unsigned)f2bfu(hi) << 16);
}
// A-operand (16x32 bf16) K-offset for VGPR v (CDNA5 ISA 7.12.2)
static __device__ __forceinline__ int aK(int lane, int v) {
  int k = (v < 4) ? (2 * v) : (16 + 2 * (v - 4));
  return k + ((lane & 16) ? 8 : 0);
}
static __device__ __forceinline__ v8f wmma_bf16(v8u a, v8u b, v8f c) {
  return __builtin_amdgcn_wmma_f32_16x16x32_bf16(
      false, __builtin_bit_cast(v16bf, a),
      false, __builtin_bit_cast(v16bf, b),
      (short)0, c, false, false);
}

// ---------- TDM: load hT[b][0:256][j0:j0+32] (bf16) -> padded LDS [256][40] ----------
#if GAT_TDM
static __device__ __forceinline__ void tdm_issue(const unsigned short* gsrc, unsigned ldsoff) {
  typedef unsigned int u32x4 __attribute__((ext_vector_type(4)));
  typedef int i32x4 __attribute__((ext_vector_type(4)));
  typedef int i32x8 __attribute__((ext_vector_type(8)));
  size_t ga = (size_t)gsrc;
  u32x4 g0;
  g0[0] = 1u;                                    // count=1 user descriptor
  g0[1] = ldsoff;                                // lds_addr
  g0[2] = (unsigned)ga;                          // global_addr[31:0]
  g0[3] = (unsigned)(ga >> 32) | (2u << 30);     // global_addr[56:32] | type=2 (image)
  i32x8 g1;
  g1[0] = (1 << 16) | (1 << 20) | (3 << 22) | (3 << 25); // data_size=2B, pad_en, every 16 dwords pad 4 dwords
  g1[1] = (int)((unsigned)NN << 16);             // tensor_dim0[15:0]=2048  (bits 79:48)
  g1[2] = (int)((unsigned)FF << 16);             // tensor_dim1[15:0]=256   (bits 111:80)
  g1[3] = (int)(32u << 16);                      // tile_dim0=32            (bits 127:112)
  g1[4] = FF;                                    // tile_dim1=256, tile_dim2=0
  g1[5] = NN;                                    // tensor_dim0_stride lo
  g1[6] = 0;                                     // stride0 hi | stride1[15:0]
  g1[7] = (int)(((size_t)NN * FF) >> 16);        // tensor_dim1_stride[47:16]
  i32x4 z4 = {0, 0, 0, 0};
  i32x8 z8 = {0, 0, 0, 0, 0, 0, 0, 0};
  // 6-arg form (clang-23 / therock HIP headers)
  __builtin_amdgcn_tensor_load_to_lds(g0, g1, z4, z4, z8, 0);
}
#endif
// fallback: cooperative copy of already-transposed slab (same LDS layout)
static __device__ __forceinline__ void stage_copy(const unsigned short* gsrc, unsigned short* dst) {
  int t = threadIdx.x;
#pragma unroll
  for (int i = 0; i < 4; ++i) {
    int q = t + 256 * i;            // 1024 chunks of 16B
    int d = q >> 2;
    int c = (q & 3) * 8;            // element offset within row
    uint4 v = *(const uint4*)(gsrc + (size_t)d * NN + c);
    *(uint4*)&dst[d * TSTR + c] = v;
  }
}

// ---------------- K0: W[f][d] (f32) -> WT[d][f] (bf16) ----------------
__global__ void gat_convert_wt(const float* __restrict__ W, unsigned short* __restrict__ WT) {
  int idx = blockIdx.x * 256 + threadIdx.x;
  int k = idx >> 8, d = idx & 255;
  WT[(size_t)d * FF + k] = f2bfu(W[idx]);
}

// ---------------- K1: h = x @ W, emit row-major h and transposed hT (bf16) ----------------
__global__ void __launch_bounds__(256) gat_h_gemm(const float* __restrict__ x,
                                                  const unsigned short* __restrict__ WT,
                                                  unsigned short* __restrict__ hbf,
                                                  unsigned short* __restrict__ hTg) {
  __shared__ unsigned short xA[16 * HSTR];
  const int t = threadIdx.x, wave = t >> 5, lane = t & 31;
  const int m = lane & 15, hi = lane >> 4;
  const int r0 = blockIdx.x * 16;
  const int d0 = wave * 32;
  v8f c0 = {}; v8f c1 = {};

  for (int k0 = 0; k0 < FF; k0 += 32) {
    {
      int mm = t >> 4, kk = (t & 15) * 2;
      const float* xp = x + (size_t)(r0 + mm) * FF + k0 + kk;
      *(unsigned*)&xA[mm * HSTR + kk] = packbf(xp[0], xp[1]);
    }
    __syncthreads();
    v8u au;
#pragma unroll
    for (int v = 0; v < 8; ++v)
      au[v] = *(const unsigned*)&xA[m * HSTR + aK(lane, v)];
    const int khalf = hi * 16;
    v8u bu0, bu1;
#pragma unroll
    for (int v = 0; v < 8; ++v) {
      int kb = k0 + aK(lane, v);
      bu0[v] = *(const unsigned*)&WT[(size_t)(d0 + m) * FF + kb];
      bu1[v] = *(const unsigned*)&WT[(size_t)(d0 + 16 + m) * FF + kb];
    }
    c0 = wmma_bf16(au, bu0, c0);
    c1 = wmma_bf16(au, bu1, c1);
    __syncthreads();
  }
#pragma unroll
  for (int v = 0; v < 8; ++v) {
    int mr = v + 8 * hi;
    int grow = r0 + mr;
    size_t row = (size_t)grow * FF;
    unsigned short b0 = f2bfu(c0[v]), b1 = f2bfu(c1[v]);
    hbf[row + d0 + m]      = b0;
    hbf[row + d0 + 16 + m] = b1;
    int bb = grow >> 11, jj = grow & (NN - 1);
    hTg[((size_t)bb * FF + d0 + m) * NN + jj]      = b0;
    hTg[((size_t)bb * FF + d0 + 16 + m) * NN + jj] = b1;
  }
}

// ---------------- K2: attention score projections ----------------
__global__ void gat_scores(const unsigned short* __restrict__ hbf,
                           const float* __restrict__ a,
                           float* __restrict__ ssrc, float* __restrict__ sdst) {
  const int wave = threadIdx.x >> 5, lane = threadIdx.x & 31;
  const int r = blockIdx.x * 8 + wave;
  const unsigned short* hr = hbf + (size_t)r * FF;
  float p0 = 0.f, p1 = 0.f;
#pragma unroll
  for (int d = lane; d < FF; d += 32) {
    float hv = bfu2f(hr[d]);
    p0 += hv * a[d];
    p1 += hv * a[FF + d];
  }
#pragma unroll
  for (int off = 16; off > 0; off >>= 1) {
    p0 += __shfl_xor(p0, off, 32);
    p1 += __shfl_xor(p1, off, 32);
  }
  if (lane == 0) { ssrc[r] = p0; sdst[r] = p1; }
}

// ---------------- K3: fused flash-GAT, M=32 rows/WG, TDM double-buffered ----------------
__global__ void __launch_bounds__(256) gat_flash(const int* __restrict__ Amask,
                                                 const unsigned short* __restrict__ hTg,
                                                 const float* __restrict__ ssrc,
                                                 const float* __restrict__ sdst,
                                                 float* __restrict__ y) {
  __shared__ unsigned short hT2[2][FF * TSTR];   // 2 x 20 KB transposed h slabs
  __shared__ unsigned short Pb[32 * HSTR];       // probs, A-operand readable
  __shared__ float mrun[32], lrun[32], sscale[32];

  const int t = threadIdx.x, wave = t >> 5, lane = t & 31;
  const int m = lane & 15, hi = lane >> 4;
  const int d0 = wave * 32;

  const int b  = blockIdx.x >> 6;            // / (NN/32)
  const int i0 = (blockIdx.x & 63) * 32;
  const size_t rowbase = (size_t)b * NN;
  const int* Ab = Amask + ((size_t)b * NN + i0) * NN;
  const unsigned short* hTb = hTg + (size_t)b * FF * NN;

  if (t < 32) { mrun[t] = -1e30f; lrun[t] = 0.f; }

  // P-phase: thread t -> rows rA=t>>4 and rB=rA+16, cols (pk, pk+1)
  const int rA = t >> 4, rB = rA + 16;
  const int pk = (t & 15) * 2;
  const float srcA = ssrc[rowbase + i0 + rA];
  const float srcB = ssrc[rowbase + i0 + rB];

  v8f c00 = {}, c01 = {}, c10 = {}, c11 = {};

  // prime buffer 0
#if GAT_TDM
  if (wave == 0) tdm_issue(hTb, (unsigned)(size_t)&hT2[0][0]);
#else
  stage_copy(hTb, &hT2[0][0]);
#endif
  __syncthreads();   // mrun/lrun init visible (TDM completion enforced below)

  for (int it = 0; it < NT; ++it) {
    const int j0 = it * 32;
    const unsigned short* hTc = &hT2[it & 1][0];

    // ---- stage next tile / wait current ----
#if GAT_TDM
    if (wave == 0) {
      if (it + 1 < NT) {
        tdm_issue(hTb + (j0 + 32), (unsigned)(size_t)&hT2[(it + 1) & 1][0]);
        __builtin_amdgcn_s_wait_tensorcnt(1);   // current slab complete (in-order)
      } else {
        __builtin_amdgcn_s_wait_tensorcnt(0);
      }
    }
#else
    if (it + 1 < NT) stage_copy(hTb + (j0 + 32), &hT2[(it + 1) & 1][0]);
#endif

    // ---- phase 1: logits + running max ----
    const int a0 = Ab[(size_t)rA * NN + j0 + pk];
    const int a1 = Ab[(size_t)rA * NN + j0 + pk + 1];
    const int b0 = Ab[(size_t)rB * NN + j0 + pk];
    const int b1 = Ab[(size_t)rB * NN + j0 + pk + 1];
    const float sd0 = sdst[rowbase + j0 + pk];
    const float sd1 = sdst[rowbase + j0 + pk + 1];
    float zA0 = srcA + sd0, zA1 = srcA + sd1, zB0 = srcB + sd0, zB1 = srcB + sd1;
    zA0 = (zA0 >= 0.f) ? zA0 : NEG_SLOPE * zA0;
    zA1 = (zA1 >= 0.f) ? zA1 : NEG_SLOPE * zA1;
    zB0 = (zB0 >= 0.f) ? zB0 : NEG_SLOPE * zB0;
    zB1 = (zB1 >= 0.f) ? zB1 : NEG_SLOPE * zB1;
    float rmaxA = fmaxf((a0 > 0) ? zA0 : -1e30f, (a1 > 0) ? zA1 : -1e30f);
    float rmaxB = fmaxf((b0 > 0) ? zB0 : -1e30f, (b1 > 0) ? zB1 : -1e30f);
#pragma unroll
    for (int off = 8; off > 0; off >>= 1) {
      rmaxA = fmaxf(rmaxA, __shfl_xor(rmaxA, off, 16));
      rmaxB = fmaxf(rmaxB, __shfl_xor(rmaxB, off, 16));
    }
    if ((t & 15) == 0) {
      float moA = mrun[rA], mnA = fmaxf(moA, rmaxA);
      sscale[rA] = __expf(moA - mnA); mrun[rA] = mnA;
      float moB = mrun[rB], mnB = fmaxf(moB, rmaxB);
      sscale[rB] = __expf(moB - mnB); mrun[rB] = mnB;
    }
    if (j0 + 32 < NN) {   // prefetch next mask tiles
      __builtin_prefetch(&Ab[(size_t)rA * NN + j0 + 32 + pk], 0, 1);
      __builtin_prefetch(&Ab[(size_t)rB * NN + j0 + 32 + pk], 0, 1);
    }
    __syncthreads();   // slab staged + m_new visible

    // ---- phase 2: probabilities + running sum ----
    const float mnA = mrun[rA], mnB = mrun[rB];
    const float pA0 = (a0 > 0) ? __expf(zA0 - mnA) : 0.f;
    const float pA1 = (a1 > 0) ? __expf(zA1 - mnA) : 0.f;
    const float pB0 = (b0 > 0) ? __expf(zB0 - mnB) : 0.f;
    const float pB1 = (b1 > 0) ? __expf(zB1 - mnB) : 0.f;
    *(unsigned*)&Pb[rA * HSTR + pk] = packbf(pA0, pA1);
    *(unsigned*)&Pb[rB * HSTR + pk] = packbf(pB0, pB1);
    float rsA = pA0 + pA1, rsB = pB0 + pB1;
#pragma unroll
    for (int off = 8; off > 0; off >>= 1) {
      rsA += __shfl_xor(rsA, off, 16);
      rsB += __shfl_xor(rsB, off, 16);
    }
    if ((t & 15) == 0) {
      lrun[rA] = lrun[rA] * sscale[rA] + rsA;
      lrun[rB] = lrun[rB] * sscale[rB] + rsB;
    }
    __syncthreads();   // P + scales visible

    // ---- phase 3: rescale + 4 WMMAs ----
#pragma unroll
    for (int v = 0; v < 8; ++v) {
      float scA = sscale[v + 8 * hi];
      float scB = sscale[16 + v + 8 * hi];
      c00[v] *= scA; c01[v] *= scA;
      c10[v] *= scB; c11[v] *= scB;
    }
    v8u auA, auB;
#pragma unroll
    for (int v = 0; v < 8; ++v) {
      int k = hi * 16 + 2 * v;
      auA[v] = *(const unsigned*)&Pb[m * HSTR + k];
      auB[v] = *(const unsigned*)&Pb[(16 + m) * HSTR + k];
    }
    const int khalf = hi * 16;
    int base0 = (d0 + m) * TSTR + khalf;
    int base1 = (d0 + 16 + m) * TSTR + khalf;
    uint4 q0 = *(const uint4*)&hTc[base0];
    uint4 q1 = *(const uint4*)&hTc[base0 + 8];
    uint4 q2 = *(const uint4*)&hTc[base1];
    uint4 q3 = *(const uint4*)&hTc[base1 + 8];
    v8u bu0 = {q0.x, q0.y, q0.z, q0.w, q1.x, q1.y, q1.z, q1.w};
    v8u bu1 = {q2.x, q2.y, q2.z, q2.w, q3.x, q3.y, q3.z, q3.w};
    c00 = wmma_bf16(auA, bu0, c00);
    c01 = wmma_bf16(auA, bu1, c01);
    c10 = wmma_bf16(auB, bu0, c10);
    c11 = wmma_bf16(auB, bu1, c11);
    __syncthreads();   // all LDS reads done before buffers are reused
  }

  // ---- epilogue: y = acc / l ----
#pragma unroll
  for (int v = 0; v < 8; ++v) {
    int mrA = v + 8 * hi, mrB = 16 + mrA;
    float invA = 1.f / lrun[mrA];
    float invB = 1.f / lrun[mrB];
    size_t rowA = (rowbase + i0 + mrA) * (size_t)FF;
    size_t rowB = (rowbase + i0 + mrB) * (size_t)FF;
    y[rowA + d0 + m]      = c00[v] * invA;
    y[rowA + d0 + 16 + m] = c01[v] * invA;
    y[rowB + d0 + m]      = c10[v] * invB;
    y[rowB + d0 + 16 + m] = c11[v] * invB;
  }
}

// ---------------- launcher ----------------
extern "C" void kernel_launch(void* const* d_in, const int* in_sizes, int n_in,
                              void* d_out, int out_size, void* d_ws, size_t ws_size,
                              hipStream_t stream) {
  const float* x  = (const float*)d_in[0];
  const int*   A  = (const int*)d_in[1];
  const float* W  = (const float*)d_in[2];
  const float* av = (const float*)d_in[3];
  float* y = (float*)d_out;

  char* ws = (char*)d_ws;
  unsigned short* WT  = (unsigned short*)ws;                              // 128 KB
  unsigned short* hbf = (unsigned short*)(ws + 131072);                   // 8 MB
  unsigned short* hTg = (unsigned short*)(ws + 131072 + (size_t)BN * FF * 2);        // 8 MB
  float* ssrc = (float*)(ws + 131072 + 2 * (size_t)BN * FF * 2);
  float* sdst = ssrc + BN;

  gat_convert_wt<<<FF, 256, 0, stream>>>(W, WT);
  gat_h_gemm<<<BN / 16, 256, 0, stream>>>(x, WT, hbf, hTg);
  gat_scores<<<BN / 8, 256, 0, stream>>>(hbf, av, ssrc, sdst);
  gat_flash<<<BATCH * (NN / 32), 256, 0, stream>>>(A, hTg, ssrc, sdst, y);
}
